// Flow_58239756534099
// MI455X (gfx1250) — matrix operand order repaired and checked by reading, not compiled, not run on hardware
//
#include <hip/hip_runtime.h>
#include <math.h>

typedef _Float16 v16h __attribute__((ext_vector_type(16)));
typedef _Float16 v8h  __attribute__((ext_vector_type(8)));
typedef float    v8f  __attribute__((ext_vector_type(8)));

#define BATCH  32
#define SEQ    4096
#define ROWS   (BATCH*SEQ)
#define HID    256
#define CTXD   64
#define TILE_M 128
#define LDA    264   // padded f16 row stride in LDS (264*2B = 528B -> bank-conflict-free, 16B aligned)

// ---------------------------------------------------------------------------
// Prep kernel 1: convert f32 weights W[Nreal][Kreal] (row-major, as in the
// reference: W[o][k]) into f16 WMMA B-matrix fragments.
// B fragment in the contiguous k order used here (16-bit, 32x16): lane holds column n = nt*16 + (lane&15),
// element i holds K = kt*32 + (lane>=16 ? 16 : 0) + i. One lane = 16 f16 =
// 32 contiguous bytes, so the consumer does a single 32B global load.
// ---------------------------------------------------------------------------
__global__ __launch_bounds__(256)
void make_frags(const float* __restrict__ W, _Float16* __restrict__ dst,
                int Kreal, int Ktiles, int Ntiles, int Nreal)
{
    int t = blockIdx.x * blockDim.x + threadIdx.x;
    if (t >= Ktiles * Ntiles * 32) return;
    int lane = t & 31;
    int ktnt = t >> 5;
    int kt = ktnt / Ntiles;
    int nt = ktnt % Ntiles;
    int n  = nt * 16 + (lane & 15);
    int k0 = kt * 32 + ((lane >= 16) ? 16 : 0);
    _Float16* o = dst + (size_t)((kt * Ntiles + nt) * 32 + lane) * 16;
    #pragma unroll
    for (int i = 0; i < 16; ++i) {
        int k = k0 + i;
        float v = (n < Nreal && k < Kreal) ? W[(size_t)n * Kreal + k] : 0.f;
        o[i] = (_Float16)v;
    }
}

// ---------------------------------------------------------------------------
// Prep kernel 2: per-batch hyper terms folded into scale/shift:
//   out = (dot + b) * sigmoid(Wg z + bg) + Wb z  ==  dot*scale + shift
//   scale = sigmoid(Wg z + bg); shift = b*scale + Wb z
// ---------------------------------------------------------------------------
__global__ __launch_bounds__(256)
void make_tables(const float* __restrict__ z,  const float* __restrict__ Wg,
                 const float* __restrict__ bg, const float* __restrict__ Wb,
                 const float* __restrict__ bmain,
                 float* __restrict__ sc, float* __restrict__ sh,
                 int DOreal, int DOpad)
{
    int t = blockIdx.x * blockDim.x + threadIdx.x;
    if (t >= 32 * DOpad) return;
    int bb = t / DOpad, o = t % DOpad;
    float s = 0.f, h = 0.f;
    if (o < DOreal) {
        float g = bg[o], bz = 0.f;
        const float* zr = z + (size_t)bb * CTXD;
        for (int k = 0; k < CTXD; ++k) {
            g  += zr[k] * Wg[(size_t)o * CTXD + k];
            bz += zr[k] * Wb[(size_t)o * CTXD + k];
        }
        float gate = 1.f / (1.f + expf(-g));
        s = gate;
        h = bmain[o] * gate + bz;
    }
    sc[t] = s; sh[t] = h;
}

__device__ __forceinline__ v8f wmma_f16(v16h a, v16h b, v8f c) {
    return __builtin_amdgcn_wmma_f32_16x16x32_f16(false, a, false, b,
                                                  (short)0, c, false, false);
}

// Build A-matrix fragment in contiguous k order (16-bit 16x32): lane row m = lane&15,
// element i: K = kbase + i, kbase = (lane>=16)?16:0.
// Two contiguous 16B LDS loads per fragment.
__device__ __forceinline__ v16h load_afrag(const _Float16* ap) {
    v8h lo = *(const v8h*)(ap);
    v8h hi = *(const v8h*)(ap + 8);
    return __builtin_shufflevector(lo, hi, 0,1,2,3,4,5,6,7,8,9,10,11,12,13,14,15);
}

// ---------------------------------------------------------------------------
// Fused branch kernel: 256 threads = 8 wave32, 128 rows per block (all rows
// in one block share the same batch index since 128 | 4096). Each wave owns a
// 16-row strip; activations live in LDS for the whole 4-layer pipeline.
// ---------------------------------------------------------------------------
template<int DIN>
__global__ __launch_bounds__(256)
void branch_kernel(const float* __restrict__ xin, const float* __restrict__ pstd,
                   const float* __restrict__ w0,
                   const float* __restrict__ sc0, const float* __restrict__ sh0,
                   const _Float16* __restrict__ wf1,
                   const float* __restrict__ sc1, const float* __restrict__ sh1,
                   const _Float16* __restrict__ wf2,
                   const float* __restrict__ sc2, const float* __restrict__ sh2,
                   const _Float16* __restrict__ wf3,
                   const float* __restrict__ sc3, const float* __restrict__ sh3,
                   float* __restrict__ out, int colOff)
{
    __shared__ __align__(16) _Float16 act[TILE_M * LDA];

    const int tid  = threadIdx.x;
    const int wave = tid >> 5, lane = tid & 31;
    const int row0 = blockIdx.x * TILE_M;
    const int bb   = row0 / SEQ;            // whole tile is in one batch
    const int wr0  = wave * 16;
    const int mofs = (lane >= 16) ? 8 : 0;  // C/D row offset for upper half-wave
    const int kofs = (lane >= 16) ? 16 : 0; // A K-offset for upper half-wave

    // ---- layer 0: K = 2 or 3, do it on VALU, write f16 activations ----
    {
        const float* scp = sc0 + (size_t)bb * HID;
        const float* shp = sh0 + (size_t)bb * HID;
        for (int m = 0; m < 16; ++m) {
            int row = row0 + wr0 + m;
            float i0 = xin[(size_t)row * 3 + 0];
            float i1 = (DIN == 3) ? xin[(size_t)row * 3 + 1] : pstd[row];
            float i2 = (DIN == 3) ? pstd[row] : 0.f;
            for (int c = lane; c < HID; c += 32) {
                const float* wr = w0 + (size_t)c * DIN;
                float d = wr[0] * i0 + wr[1] * i1;
                if (DIN == 3) d += wr[2] * i2;
                float v = tanhf(d * scp[c] + shp[c]);
                act[(wr0 + m) * LDA + c] = (_Float16)v;
            }
        }
    }
    __syncthreads();

    // ---- layers 1 & 2: per wave, 16x256 = A(16x256) x W^T(256x256) via WMMA ----
    #pragma unroll
    for (int layer = 0; layer < 2; ++layer) {
        const _Float16* wf = layer ? wf2 : wf1;
        const float* scp = (layer ? sc2 : sc1) + (size_t)bb * HID;
        const float* shp = (layer ? sh2 : sh1) + (size_t)bb * HID;

        v8f zero = {};
        v8f acc[16];
        #pragma unroll
        for (int nt = 0; nt < 16; ++nt) acc[nt] = zero;

        #pragma unroll
        for (int kt = 0; kt < 8; ++kt) {
            const _Float16* ap = &act[(wr0 + (lane & 15)) * LDA + kt * 32 + kofs];
            v16h a = load_afrag(ap);
            const _Float16* bp = wf + (size_t)(kt * 16 * 32 + lane) * 16;
            #pragma unroll
            for (int nt = 0; nt < 16; ++nt) {
                v16h b = *(const v16h*)(bp + (size_t)nt * 32 * 16);
                acc[nt] = wmma_f16(a, b, acc[nt]);
            }
        }
        __syncthreads();   // all A reads done before overwriting the tile
        #pragma unroll
        for (int nt = 0; nt < 16; ++nt) {
            int col = nt * 16 + (lane & 15);
            float s = scp[col], h = shp[col];
            #pragma unroll
            for (int r = 0; r < 8; ++r) {
                float v = tanhf(acc[nt][r] * s + h);
                act[(wr0 + r + mofs) * LDA + col] = (_Float16)v;
            }
        }
        __syncthreads();
    }

    // ---- layer 3: 256 -> 2 (N padded to 16 in the weight fragments) ----
    {
        v8f acc3 = {};
        #pragma unroll
        for (int kt = 0; kt < 8; ++kt) {
            const _Float16* ap = &act[(wr0 + (lane & 15)) * LDA + kt * 32 + kofs];
            v16h a = load_afrag(ap);
            v16h b = *(const v16h*)(wf3 + (size_t)(kt * 32 + lane) * 16);
            acc3 = wmma_f16(a, b, acc3);
        }
        int col = lane & 15;               // output channel (only 0,1 are real)
        if (col < 2) {
            float s = sc3[bb * 16 + col], h = sh3[bb * 16 + col];
            // channel 0 -> log_s plane, channel 1 -> b plane; colOff = branch slot
            float* plane = out + (size_t)col * ((size_t)ROWS * 2);
            #pragma unroll
            for (int r = 0; r < 8; ++r) {
                int row = row0 + wr0 + r + mofs;
                plane[(size_t)row * 2 + colOff] = acc3[r] * s + h;
            }
        }
    }
}

// ---------------------------------------------------------------------------
extern "C" void kernel_launch(void* const* d_in, const int* in_sizes, int n_in,
                              void* d_out, int out_size, void* d_ws, size_t ws_size,
                              hipStream_t stream)
{
    (void)in_sizes; (void)n_in; (void)out_size; (void)ws_size;

    const float* xin  = (const float*)d_in[0];
    const float* pstd = (const float*)d_in[1];
    const float* z    = (const float*)d_in[2];
    // params1 layers at d_in[3 + 5*i + {W,b,Wb,Wg,bg}], params2 at d_in[23 + ...]

    char*  ws  = (char*)d_ws;
    size_t off = 0;
    auto alloc = [&](size_t bytes) -> void* {
        void* p = ws + off;
        off = (off + bytes + 255) & ~(size_t)255;
        return p;
    };

    for (int br = 0; br < 2; ++br) {
        const int base = 3 + br * 20;
        const float* W[4]; const float* Bm[4]; const float* Wb[4];
        const float* Wg[4]; const float* Bg[4];
        for (int i = 0; i < 4; ++i) {
            W[i]  = (const float*)d_in[base + 5*i + 0];
            Bm[i] = (const float*)d_in[base + 5*i + 1];
            Wb[i] = (const float*)d_in[base + 5*i + 2];
            Wg[i] = (const float*)d_in[base + 5*i + 3];
            Bg[i] = (const float*)d_in[base + 5*i + 4];
        }

        _Float16* wf1 = (_Float16*)alloc((size_t)8*16*32*16 * 2);
        _Float16* wf2 = (_Float16*)alloc((size_t)8*16*32*16 * 2);
        _Float16* wf3 = (_Float16*)alloc((size_t)8* 1*32*16 * 2);
        float* sc[4]; float* sh[4];
        for (int i = 0; i < 3; ++i) {
            sc[i] = (float*)alloc((size_t)32*HID * 4);
            sh[i] = (float*)alloc((size_t)32*HID * 4);
        }
        sc[3] = (float*)alloc((size_t)32*16 * 4);
        sh[3] = (float*)alloc((size_t)32*16 * 4);

        // weight fragments (layers 1,2 full 256x256; layer 3 padded 2->16)
        make_frags<<<(8*16*32 + 255)/256, 256, 0, stream>>>(W[1], wf1, HID, 8, 16, HID);
        make_frags<<<(8*16*32 + 255)/256, 256, 0, stream>>>(W[2], wf2, HID, 8, 16, HID);
        make_frags<<<(8* 1*32 + 255)/256, 256, 0, stream>>>(W[3], wf3, HID, 8,  1, 2);

        // per-batch scale/shift tables
        for (int i = 0; i < 3; ++i)
            make_tables<<<(32*HID + 255)/256, 256, 0, stream>>>(
                z, Wg[i], Bg[i], Wb[i], Bm[i], sc[i], sh[i], HID, HID);
        make_tables<<<(32*16 + 255)/256, 256, 0, stream>>>(
            z, Wg[3], Bg[3], Wb[3], Bm[3], sc[3], sh[3], 2, 16);

        // fused 4-layer branch MLP
        if (br == 0) {
            branch_kernel<2><<<ROWS/TILE_M, 256, 0, stream>>>(
                xin, pstd, W[0], sc[0], sh[0], wf1, sc[1], sh[1],
                wf2, sc[2], sh[2], wf3, sc[3], sh[3], (float*)d_out, 0);
        } else {
            branch_kernel<3><<<ROWS/TILE_M, 256, 0, stream>>>(
                xin, pstd, W[0], sc[0], sh[0], wf1, sc[1], sh[1],
                wf2, sc[2], sh[2], wf3, sc[3], sh[3], (float*)d_out, 1);
        }
    }
}
